// GraphSAGEConvolution_64132451664570
// MI455X (gfx1250) — matrix operand order repaired and checked by reading, not compiled, not run on hardware
//
#include <hip/hip_runtime.h>
#include <stdint.h>

typedef __attribute__((ext_vector_type(16))) __bf16 v16bf;
typedef __attribute__((ext_vector_type(8)))  float  v8f;

#define N_NODES 50000
#define F_IN    128
#define F_OUT   128
#define NPB     32          // nodes per block (2 WMMA M-subtiles)

__device__ __forceinline__ unsigned int f32bits(float x) {
    union { float f; unsigned int u; } c; c.f = x; return c.u;
}
__device__ __forceinline__ float bitsf32(unsigned int u) {
    union { float f; unsigned int u; } c; c.u = u; return c.f;
}
// pack two floats into one dword of 2 truncated bf16 (hi parts)
__device__ __forceinline__ unsigned int pack_hi(float v0, float v1) {
    return (f32bits(v1) & 0xFFFF0000u) | (f32bits(v0) >> 16);
}
// residuals after removing the hi bf16 part
__device__ __forceinline__ unsigned int pack_lo(float v0, float v1) {
    float r0 = v0 - bitsf32(f32bits(v0) & 0xFFFF0000u);
    float r1 = v1 - bitsf32(f32bits(v1) & 0xFFFF0000u);
    return (f32bits(r1) & 0xFFFF0000u) | (f32bits(r0) >> 16);
}

union FragBF { v16bf v; unsigned int u[8]; };

// A-fragment index for value pair (k, k+1) of row r (within block), k even.
// Layout [m][kc][lane][j] -> each lane's 8 dwords contiguous (ds_load_b128 x2).
// ISA A layout (16x32 bf16): lanes 0-15: K 0-7 & 16-23; lanes 16-31: +8.
__device__ __forceinline__ int afrag_index(int r, int k) {
    int m      = r >> 4;
    int lane15 = r & 15;
    int kc     = k >> 5;
    int kk     = k & 31;
    int lane16 = (kk >> 3) & 1;                       // K+8 half -> lanes 16-31
    int j      = ((kk & 16) >> 2) + ((kk & 7) >> 1);  // VGPR within fragment
    int lane   = lane15 | (lane16 << 4);
    return (((m * 8 + kc) * 32) + lane) * 8 + j;
}

// ---------------------------------------------------------------------------
// Pre-swizzle weight [256][128] into WMMA B-fragment order (hi/lo bf16 split).
// Layout [ct][kc][lane][j] -> wave loads 2x global_load_b128 per fragment.
// B layout (16x16x32 bf16): lanes 0-15 K 0-7 & 16-23 (VGPR j<4: K=2j,2j+1;
// j>=4: K=16+2(j-4)), lanes 16-31: +8; column N = lane & 15.
// ---------------------------------------------------------------------------
__global__ void prep_wfrag(const float* __restrict__ W,
                           unsigned int* __restrict__ whi,
                           unsigned int* __restrict__ wlo) {
    int idx  = blockIdx.x * blockDim.x + threadIdx.x;   // 0..16383
    int j    = idx & 7;
    int lane = (idx >> 3) & 31;
    int kc   = (idx >> 8) & 7;
    int ct   = (idx >> 11) & 7;
    int n    = ct * 16 + (lane & 15);
    int k0   = kc * 32 + ((j & 4) ? 16 : 0) + 2 * (j & 3) + ((lane & 16) ? 8 : 0);
    float w0 = W[k0 * F_OUT + n];
    float w1 = W[(k0 + 1) * F_OUT + n];
    whi[idx] = pack_hi(w0, w1);
    wlo[idx] = pack_lo(w0, w1);
}

// ---------------------------------------------------------------------------
// Fused: binary-search segment bounds + SpMM packing A-frags into LDS +
// split-bf16 WMMA GEMM.  Phase C is pure ds_load_b128 + v_wmma.
// ---------------------------------------------------------------------------
__global__ __launch_bounds__(256)
void gsage_fused(const float* __restrict__ x,
                 const int*   __restrict__ adj_row,
                 const int*   __restrict__ adj_col,
                 const float* __restrict__ adj_val,
                 const float* __restrict__ bias,
                 const unsigned int* __restrict__ whi,
                 const unsigned int* __restrict__ wlo,
                 float* __restrict__ out,
                 int n_edges) {
    __shared__ unsigned int ahi[2 * 8 * 32 * 8];   // 4096 dwords = 16 KB
    __shared__ unsigned int alo[2 * 8 * 32 * 8];   // 16 KB
    __shared__ int bounds[NPB + 1];

    const int tid       = threadIdx.x;
    const int node_base = blockIdx.x * NPB;

    // ---- Phase A: edge ranges via lower_bound on sorted adj_row ----
    if (tid <= NPB) {
        int target = node_base + tid;
        int lo = 0, hi = n_edges;
        while (lo < hi) {
            int mid = (lo + hi) >> 1;
            if (adj_row[mid] < target) lo = mid + 1; else hi = mid;
        }
        bounds[tid] = lo;
    }
    __syncthreads();

    // ---- Phase B: x copy + deterministic segment-sum, packed straight into
    //      split-bf16 A fragments (conversion done ONCE, not per wave) ----
    {
        const int f2 = tid & 63;                  // feature-pair index 0..63
        const int kx = 2 * f2;                    // h column of x half
        const int kn = 128 + 2 * f2;              // h column of nei half
        for (int r = (tid >> 6); r < NPB; r += 4) {
            int   node = node_base + r;
            float xv0 = 0.f, xv1 = 0.f, ac0 = 0.f, ac1 = 0.f;
            if (node < N_NODES) {
                const float* xr = &x[node * F_IN + kx];
                xv0 = xr[0]; xv1 = xr[1];         // merged b64, coalesced
                const int e1 = bounds[r + 1];
                for (int e = bounds[r]; e < e1; ++e) {
                    int   c = adj_col[e];         // uniform across wave
                    float v = adj_val[e];
                    const float* xc = &x[c * F_IN + kx];
                    ac0 += v * xc[0];             // 512B coalesced, L2-resident
                    ac1 += v * xc[1];
                }
            }
            int ix = afrag_index(r, kx);
            ahi[ix] = pack_hi(xv0, xv1);
            alo[ix] = pack_lo(xv0, xv1);
            int in_ = afrag_index(r, kn);
            ahi[in_] = pack_hi(ac0, ac1);
            alo[in_] = pack_lo(ac0, ac1);
        }
    }
    __syncthreads();

    // ---- Phase C: h[32x256] @ W[256x128] + bias via split-bf16 WMMA ----
    const int wave = tid >> 5;                    // 0..7 = 16-col output tile
    const int lane = tid & 31;
    const int col  = (wave << 4) + (lane & 15);
    const float bv = bias[col];

    v8f c0 = {};
    v8f c1 = {};

    for (int kc = 0; kc < 8; ++kc) {
        // B fragments: 2x global_load_b128 each (1 KB contiguous per wave)
        FragBF bh, bl;
        const int wb = (((wave * 8 + kc) * 32) + lane) * 8;
#pragma unroll
        for (int j = 0; j < 8; ++j) {
            bh.u[j] = whi[wb + j];
            bl.u[j] = wlo[wb + j];
        }
#pragma unroll
        for (int m = 0; m < 2; ++m) {
            // A fragments: 2x ds_load_b128 each, zero conversion VALU
            FragBF ah, al;
            const int ab = (((m * 8 + kc) * 32) + lane) * 8;
#pragma unroll
            for (int j = 0; j < 8; ++j) {
                ah.u[j] = ahi[ab + j];
                al.u[j] = alo[ab + j];
            }
            v8f c = (m == 0) ? c0 : c1;
            // C += Ah*Bh + Ah*Bl + Al*Bh  (fp32 accumulate, ~2^-16 rel error)
            c = __builtin_amdgcn_wmma_f32_16x16x32_bf16(false, ah.v, false, bh.v,
                                                        (short)0, c, false, false);
            c = __builtin_amdgcn_wmma_f32_16x16x32_bf16(false, ah.v, false, bl.v,
                                                        (short)0, c, false, false);
            c = __builtin_amdgcn_wmma_f32_16x16x32_bf16(false, al.v, false, bh.v,
                                                        (short)0, c, false, false);
            if (m == 0) c0 = c; else c1 = c;
        }
    }

    // ---- bias + coalesced stores (C/D layout: lane 0-15 M=v, 16-31 M=v+8) ----
#pragma unroll
    for (int m = 0; m < 2; ++m) {
        const v8f c = (m == 0) ? c0 : c1;
#pragma unroll
        for (int v = 0; v < 8; ++v) {
            int row = node_base + m * 16 + v + ((lane & 16) ? 8 : 0);
            if (row < N_NODES) out[row * F_OUT + col] = c[v] + bv;
        }
    }
}

// ---------------------------------------------------------------------------
extern "C" void kernel_launch(void* const* d_in, const int* in_sizes, int n_in,
                              void* d_out, int out_size, void* d_ws, size_t ws_size,
                              hipStream_t stream) {
    const float* x       = (const float*)d_in[0];
    const int*   adj_row = (const int*)  d_in[1];
    const int*   adj_col = (const int*)  d_in[2];
    const float* adj_val = (const float*)d_in[3];
    const float* weight  = (const float*)d_in[4];
    const float* bias    = (const float*)d_in[5];
    float*       out     = (float*)d_out;

    unsigned int* whi = (unsigned int*)d_ws;       // 16384 dwords = 64 KB
    unsigned int* wlo = whi + 8 * 8 * 32 * 8;      // +64 KB
    const int n_edges = in_sizes[1];

    prep_wfrag<<<64, 256, 0, stream>>>(weight, whi, wlo);

    dim3 grid((N_NODES + NPB - 1) / NPB);          // 1563 blocks
    gsage_fused<<<grid, 256, 0, stream>>>(x, adj_row, adj_col, adj_val, bias,
                                          whi, wlo, out, n_edges);
}
